// KernelAxialMultiAttention_223338300232
// MI455X (gfx1250) — matrix operand order repaired and checked by reading, not compiled, not run on hardware
//
#include <hip/hip_runtime.h>

typedef __attribute__((ext_vector_type(16))) __bf16 v16bf;
typedef __attribute__((ext_vector_type(8)))  __bf16 bf16x8;
typedef __attribute__((ext_vector_type(4)))  __bf16 bf16x4;
typedef __attribute__((ext_vector_type(8)))  float  v8f;
typedef __attribute__((ext_vector_type(2)))  float  v2f;

union AF { v16bf v; bf16x8 h[2]; };

#define TC 768    // channels
#define TT 1024   // sequence length
#define NB 64     // Bs*M batch rows
#define NH 12     // heads
#define HD 64     // head dim
#define XS 776    // LDS x-tile row stride in bf16 elements (16B aligned, padded)

// Native bf16 convert (RNE in hardware: v_cvt_bf16_f32 / v_cvt_pk_bf16_f32)
__device__ __forceinline__ __bf16 f2bf(float f) { return (__bf16)f; }
__device__ __forceinline__ unsigned short f2bfbits(float f) {
  return __builtin_bit_cast(unsigned short, (__bf16)f);
}

// Async global->LDS copy (16B per lane), tracked by ASYNCcnt  [CDNA5]
__device__ __forceinline__ void async_copy16(unsigned lds_off, const void* g) {
  asm volatile("global_load_async_to_lds_b128 %0, %1, off"
               :: "v"(lds_off), "v"((unsigned long long)(uintptr_t)g)
               : "memory");
}
__device__ __forceinline__ void async_wait0() {
  asm volatile("s_wait_asynccnt 0x0" ::: "memory");
}

// Stage a 64-row x-tile (f32 global -> bf16 LDS), 4 elems/thread/iter.
__device__ __forceinline__ void stage_x(const float* __restrict__ xsrc,
                                        __bf16* __restrict__ xs, int tid) {
  const float4* x4 = (const float4*)xsrc;   // 192 float4 per row
  for (int i = tid; i < 64 * 192; i += 256) {
    int r = i / 192, c4 = i - r * 192;
    float4 v = x4[(size_t)r * 192 + c4];
    bf16x4 o = {f2bf(v.x), f2bf(v.y), f2bf(v.z), f2bf(v.w)};
    *(bf16x4*)(xs + r * XS + c4 * 4) = o;
  }
}

// ---------------------------------------------------------------------------
__global__ void k_zero(float* __restrict__ p, int nfloats) {
  int i = blockIdx.x * 256 + threadIdx.x;
  if (i < nfloats) p[i] = 0.f;
}

// ---------------------------------------------------------------------------
// Pack wq^T / wk^T (f32 row-major [C][C]) into bf16 WMMA B-fragment layout.
// Fragment f = ct*24 + kt covers B[kt*32 .. +32)[ct*16 .. +16).
// Lane l holds K = (pos>>3)*16 + (l>>4)*8 + (pos&7) for its column n = l&15.
// ---------------------------------------------------------------------------
__global__ void k_pack(const float* __restrict__ wq, const float* __restrict__ wk,
                       unsigned short* __restrict__ wqp,
                       unsigned short* __restrict__ wkp) {
  int idx = blockIdx.x * 256 + threadIdx.x;
  const int NW = TC * TC;
  if (idx >= 2 * NW) return;
  const float* w = (idx < NW) ? wq : wk;
  unsigned short* o = (idx < NW) ? wqp : wkp;
  int e = (idx < NW) ? idx : idx - NW;
  int f = e >> 9, r = e & 511;
  int lane = r >> 4, pos = r & 15;
  int kt = f % 24, ct = f / 24;
  int k = ((pos >> 3) << 4) + ((lane >> 4) << 3) + (pos & 7);
  int nn = lane & 15;
  o[e] = f2bfbits(w[(size_t)(ct * 16 + nn) * TC + kt * 32 + k]);
}

// ---------------------------------------------------------------------------
// GEMM core: single kt sweep over ALL 6 column chunks (24 acc tiles live).
// Per kt: 8 ds_load_b128 (A, read once) + 12 global_load_b128 (B, exactly
// once from L2, double-buffered) -> 24 WMMAs.  All addresses are immediates
// off hoisted per-m / per-chunk base pointers.
#define GEMM_SWEEP(WPACK)                                                      \
  const __bf16* am[4];                                                         \
  _Pragma("unroll")                                                            \
  for (int m = 0; m < 4; ++m)                                                  \
    am[m] = xs + (m * 16 + (lane & 15)) * XS + (lane >> 4) * 8;                \
  const char* bbc[6];                                                          \
  _Pragma("unroll")                                                            \
  for (int c = 0; c < 6; ++c)                                                  \
    bbc[c] = (const char*)(WPACK) + (c * 8 + wave) * 24576 + lane * 32;        \
  v8f acc[6][4] = {};                                                          \
  AF a[4];                                                                     \
  v16bf bbuf[2][6];                                                            \
  _Pragma("unroll")                                                            \
  for (int c = 0; c < 6; ++c) bbuf[0][c] = *(const v16bf*)(bbc[c]);            \
  _Pragma("unroll 2")                                                          \
  for (int kt = 0; kt < 24; ++kt) {                                            \
    const int cur = kt & 1;                                                    \
    _Pragma("unroll")                                                          \
    for (int m = 0; m < 4; ++m) {                                              \
      a[m].h[0] = *(const bf16x8*)(am[m] + kt * 32);                           \
      a[m].h[1] = *(const bf16x8*)(am[m] + kt * 32 + 16);                      \
    }                                                                          \
    if (kt + 1 < 24) {                                                         \
      _Pragma("unroll")                                                        \
      for (int c = 0; c < 6; ++c)                                              \
        bbuf[cur ^ 1][c] = *(const v16bf*)(bbc[c] + (kt + 1) * 1024);          \
    }                                                                          \
    _Pragma("unroll")                                                          \
    for (int c = 0; c < 6; ++c)                                                \
      _Pragma("unroll")                                                        \
      for (int m = 0; m < 4; ++m)                                              \
        acc[c][m] = __builtin_amdgcn_wmma_f32_16x16x32_bf16(                   \
            false, a[m].v, false, bbuf[cur][c], (short)0, acc[c][m], false,    \
            false);                                                            \
  }

// ---------------------------------------------------------------------------
// Pass 1: K = elu(x@wk^T + bk)+1, fully fused reductions; K never hits memory.
// Grid (64 bm, 16 t-tiles), 256 threads = 8 waves.
// Wave w owns column tile ct = c*8 + w (head = c*2 + (w>>2)) for all 64 rows.
// ---------------------------------------------------------------------------
__global__ __launch_bounds__(256)
void k_pass1(const float* __restrict__ x, const __bf16* __restrict__ wkp,
             const float* __restrict__ bk, float* __restrict__ kr,
             float* __restrict__ Kr, float* __restrict__ ksum) {
  extern __shared__ char smem[];
  __bf16* xs = (__bf16*)smem;                    // 64 x XS bf16 tile of x
  float* krLDS = (float*)(smem + 64 * XS * 2);   // [64][12] row sums
  const int bm = blockIdx.x;
  const int t0 = blockIdx.y * 64;
  const int tid = threadIdx.x;

  stage_x(x + ((size_t)bm * TT + t0) * TC, xs, tid);
  if (tid < 64 * NH) krLDS[tid] = 0.f;
  __syncthreads();

  const int wave = tid >> 5, lane = tid & 31;
  GEMM_SWEEP(wkp)

  // epilogue per chunk: bias + elu + 1; ksum atomics; kr LDS-atomic merge
#pragma unroll
  for (int c = 0; c < 6; ++c) {
    const int head = c * 2 + (wave >> 2);
    const int ct = c * 8 + wave;
    const int dbase = (wave & 3) * 16;
    const float bias = bk[ct * 16 + (lane & 15)];
#pragma unroll
    for (int m = 0; m < 4; ++m)
#pragma unroll
      for (int r = 0; r < 8; ++r) {
        float v = acc[c][m][r] + bias;
        acc[c][m][r] = v > 0.f ? v + 1.f : __expf(v);
      }
    float cs = 0.f;
#pragma unroll
    for (int m = 0; m < 4; ++m)
#pragma unroll
      for (int r = 0; r < 8; ++r) cs += acc[c][m][r];
    cs += __shfl_xor(cs, 16);
    if (lane < 16)
      atomicAdd(&ksum[((size_t)bm * NH + head) * HD + dbase + lane], cs);
#pragma unroll
    for (int m = 0; m < 4; ++m)
#pragma unroll
      for (int r = 0; r < 8; ++r) {
        float v = acc[c][m][r];
        v += __shfl_xor(v, 1); v += __shfl_xor(v, 2);
        v += __shfl_xor(v, 4); v += __shfl_xor(v, 8);
        if ((lane & 15) == 0)
          atomicAdd(&krLDS[(m * 16 + (lane >> 4) * 8 + r) * NH + head], v);
      }
  }
  __syncthreads();
  // flush kr to global; tree-reduce over t for Kr
  if (tid < 64 * NH) {
    float v = krLDS[tid];
    int t = tid / NH, n = tid - t * NH;
    kr[((size_t)bm * NH + n) * TT + t0 + t] = v;
  }
#pragma unroll
  for (int st = 32; st >= 1; st >>= 1) {
    __syncthreads();
    if (tid < st * NH) krLDS[tid] += krLDS[tid + st * NH];
  }
  __syncthreads();
  if (tid < NH) atomicAdd(&Kr[bm * NH + tid], krLDS[tid]);
}

// ---------------------------------------------------------------------------
// xk[bm][n][c] = sum_t kr[bm][n][t] * x[bm][t][c]   (V is never materialized)
// ---------------------------------------------------------------------------
__global__ __launch_bounds__(128)
void k_xk(const float* __restrict__ x, const float* __restrict__ kr,
          float* __restrict__ xk) {
  __shared__ float krs[NH][128];
  const int bm = blockIdx.x;
  const int c = blockIdx.y * 128 + threadIdx.x;
  float acc[NH];
#pragma unroll
  for (int n = 0; n < NH; ++n) acc[n] = 0.f;
  for (int tc = 0; tc < TT; tc += 128) {
    __syncthreads();
    for (int i = threadIdx.x; i < NH * 128; i += 128) {
      int n = i >> 7, t = i & 127;
      krs[n][t] = kr[((size_t)bm * NH + n) * TT + tc + t];
    }
    __syncthreads();
    const float* xp = x + ((size_t)bm * TT + tc) * TC + c;
    for (int t = 0; t < 128; ++t) {
      float xv = xp[(size_t)t * TC];
#pragma unroll
      for (int n = 0; n < NH; ++n) acc[n] += krs[n][t] * xv;
    }
  }
#pragma unroll
  for (int n = 0; n < NH; ++n)
    xk[((size_t)bm * NH + n) * TC + c] = acc[n];
}

// ---------------------------------------------------------------------------
// s[bm][n][e] = wv_n[e,:] . xk[bm][n][:] + bv[n*64+e] * Kr[bm][n]
// ---------------------------------------------------------------------------
__global__ __launch_bounds__(256)
void k_s(const float* __restrict__ wv, const float* __restrict__ bv,
         const float* __restrict__ xk, const float* __restrict__ Kr,
         float* __restrict__ s) {
  const int n = blockIdx.x;
  const int e = threadIdx.x & 63;
  const int g = blockIdx.y * 4 + (threadIdx.x >> 6);  // 0..15
  const int bm0 = g * 4;
  float acc[4] = {0.f, 0.f, 0.f, 0.f};
  const float* wrow = wv + (size_t)(n * HD + e) * TC;
  for (int c = 0; c < TC; ++c) {
    float w = wrow[c];
#pragma unroll
    for (int i = 0; i < 4; ++i)
      acc[i] += w * xk[((size_t)(bm0 + i) * NH + n) * TC + c];
  }
  const float bve = bv[n * HD + e];
#pragma unroll
  for (int i = 0; i < 4; ++i)
    s[((size_t)(bm0 + i) * NH + n) * HD + e] =
        acc[i] + bve * Kr[(bm0 + i) * NH + n];
}

// ---------------------------------------------------------------------------
// u[bm][n][c] = sum_e wp[c][n*64+e] * s[bm][n][e]
// ---------------------------------------------------------------------------
__global__ __launch_bounds__(128)
void k_u(const float* __restrict__ wp, const float* __restrict__ s,
         float* __restrict__ u) {
  __shared__ float sl[NB][HD];
  const int n = blockIdx.x;
  const int c = blockIdx.y * 128 + threadIdx.x;
  for (int i = threadIdx.x; i < NB * HD; i += 128) {
    int bm = i >> 6, e = i & 63;
    sl[bm][e] = s[((size_t)bm * NH + n) * HD + e];
  }
  __syncthreads();
  float acc[NB];
#pragma unroll
  for (int bm = 0; bm < NB; ++bm) acc[bm] = 0.f;
  const float* wrow = wp + (size_t)c * TC + n * HD;
  for (int e = 0; e < HD; ++e) {
    float w = wrow[e];
#pragma unroll
    for (int bm = 0; bm < NB; ++bm) acc[bm] += sl[bm][e] * w;
  }
#pragma unroll
  for (int bm = 0; bm < NB; ++bm)
    u[((size_t)bm * NH + n) * TC + c] = acc[bm];
}

// ---------------------------------------------------------------------------
// Pass 2: q = elu(x@wq^T + bq)+1 in registers; z[t,n] = 1/(q . ksum + eps);
// out = Z[64x12] @ U[12x768] + bp via V_WMMA_F32_16X16X4_F32 (K=12 = 3x4).
// Q never hits memory.  u/bp staged with ASYNC global->LDS copies.
// ---------------------------------------------------------------------------
__global__ __launch_bounds__(256)
void k_pass2(const float* __restrict__ x, const __bf16* __restrict__ wqp,
             const float* __restrict__ bq, const float* __restrict__ ksum,
             const float* __restrict__ u, const float* __restrict__ bp,
             float* __restrict__ out) {
  extern __shared__ char smem[];
  __bf16* xs = (__bf16*)smem;                    // 64 x XS bf16
  float* us = (float*)(smem + 64 * XS * 2);      // 12 x 768
  float* bps = us + NH * TC;                     // 768
  float* zs = bps + TC;                          // 64 x 12
  const int bm = blockIdx.x;
  const int t0 = blockIdx.y * 64;
  const int tid = threadIdx.x;

  // async-stage u and bp (pure copies, 16B per lane per op)
  {
    const char* ug = (const char*)(u + (size_t)bm * NH * TC);
    unsigned ul = (unsigned)(uintptr_t)(void*)us;
    for (int i = tid * 16; i < NH * TC * 4; i += 256 * 16)
      async_copy16(ul + i, ug + i);
    const char* bg = (const char*)bp;
    unsigned bl = (unsigned)(uintptr_t)(void*)bps;
    for (int i = tid * 16; i < TC * 4; i += 256 * 16)
      async_copy16(bl + i, bg + i);
  }
  stage_x(x + ((size_t)bm * TT + t0) * TC, xs, tid);
  if (tid < 64 * NH) zs[tid] = 0.f;
  async_wait0();
  __syncthreads();

  const int wave = tid >> 5, lane = tid & 31;
  GEMM_SWEEP(wqp)

  // epilogue per chunk: (elu+1)*ksum, partial row-reduce -> zs LDS atomics
#pragma unroll
  for (int c = 0; c < 6; ++c) {
    const int head = c * 2 + (wave >> 2);
    const int ct = c * 8 + wave;
    const int dbase = (wave & 3) * 16;
    const float bias = bq[ct * 16 + (lane & 15)];
    const float ksv = ksum[((size_t)bm * NH + head) * HD + dbase + (lane & 15)];
#pragma unroll
    for (int m = 0; m < 4; ++m)
#pragma unroll
      for (int r = 0; r < 8; ++r) {
        float v = acc[c][m][r] + bias;
        acc[c][m][r] = (v > 0.f ? v + 1.f : __expf(v)) * ksv;
      }
#pragma unroll
    for (int m = 0; m < 4; ++m)
#pragma unroll
      for (int r = 0; r < 8; ++r) {
        float v = acc[c][m][r];
        v += __shfl_xor(v, 1); v += __shfl_xor(v, 2);
        v += __shfl_xor(v, 4); v += __shfl_xor(v, 8);
        if ((lane & 15) == 0)
          atomicAdd(&zs[(m * 16 + (lane >> 4) * 8 + r) * NH + head], v);
      }
  }
  __syncthreads();
  if (tid < 64 * NH) zs[tid] = 1.f / (zs[tid] + 1e-6f);
  __syncthreads();

  // out tile = Z @ U + bp with f32 WMMA (A 16x4, B 4x16, 3 k-steps)
  {
    float* op = out + ((size_t)bm * TT + t0) * TC;
    const int nlane = lane & 15;
    const int khalf = (lane >> 4) * 2;   // K pair base per lane half
    v2f za[4][3];
#pragma unroll
    for (int m = 0; m < 4; ++m) {
      const int row = m * 16 + nlane;    // A: lanes 0-15 / 16-31 both M=0..15
#pragma unroll
      for (int kk = 0; kk < 3; ++kk)
        za[m][kk] = *(const v2f*)(zs + row * NH + kk * 4 + khalf);
    }
#pragma unroll
    for (int c6 = 0; c6 < 6; ++c6) {
      const int ct = c6 * 8 + wave;
      const int col = ct * 16 + nlane;
      v2f ub[3];
#pragma unroll
      for (int kk = 0; kk < 3; ++kk) {
        ub[kk].x = us[(kk * 4 + khalf) * TC + col];
        ub[kk].y = us[(kk * 4 + khalf + 1) * TC + col];
      }
      const float bpv = bps[col];
#pragma unroll
      for (int m = 0; m < 4; ++m) {
        v8f o = {bpv, bpv, bpv, bpv, bpv, bpv, bpv, bpv};
#pragma unroll
        for (int kk = 0; kk < 3; ++kk)
          o = __builtin_amdgcn_wmma_f32_16x16x4_f32(
              false, za[m][kk], false, ub[kk], (short)0, o, false, false);
        float* ob = op + (size_t)(m * 16 + (lane >> 4) * 8) * TC + col;
#pragma unroll
        for (int r = 0; r < 8; ++r) ob[(size_t)r * TC] = o[r];
      }
    }
  }
}

// ---------------------------------------------------------------------------
extern "C" void kernel_launch(void* const* d_in, const int* in_sizes, int n_in,
                              void* d_out, int out_size, void* d_ws, size_t ws_size,
                              hipStream_t stream) {
  const float* x  = (const float*)d_in[0];
  const float* wq = (const float*)d_in[1];
  const float* bq = (const float*)d_in[2];
  const float* wk = (const float*)d_in[3];
  const float* bk = (const float*)d_in[4];
  const float* wv = (const float*)d_in[5];
  const float* bv = (const float*)d_in[6];
  const float* wp = (const float*)d_in[7];
  const float* bp = (const float*)d_in[8];
  float* out = (float*)d_out;

  char* ws = (char*)d_ws;
  __bf16* wqp = (__bf16*)(ws + 0);           // 1,179,648 B
  __bf16* wkp = (__bf16*)(ws + 1179648);     // 1,179,648 B
  float* kr   = (float*)(ws + 2359296);      // 3,145,728 B [64][12][1024]
  float* Kr   = (float*)(ws + 5505024);      //     3,072 B [64][12]
  float* ksum = (float*)(ws + 5508096);      //   196,608 B [64][12][64]
  float* xk   = (float*)(ws + 5704704);      // 2,359,296 B [64][12][768]
  float* s    = (float*)(ws + 8064000);      //   196,608 B [64][12][64]
  float* u    = (float*)(ws + 8260608);      // 2,359,296 B [64][12][768]

  k_zero<<<(49920 + 255) / 256, 256, 0, stream>>>(Kr, 49920);
  k_pack<<<(2 * TC * TC + 255) / 256, 256, 0, stream>>>(
      wq, wk, (unsigned short*)wqp, (unsigned short*)wkp);

  const size_t lds1 = (size_t)64 * XS * 2 + (size_t)64 * NH * 4;
  k_pass1<<<dim3(NB, 16), 256, lds1, stream>>>(x, wkp, bk, kr, Kr, ksum);

  k_xk<<<dim3(NB, TC / 128), 128, 0, stream>>>(x, kr, xk);
  k_s<<<dim3(NH, 4), 256, 0, stream>>>(wv, bv, xk, Kr, s);
  k_u<<<dim3(NH, TC / 128), 128, 0, stream>>>(wp, s, u);

  const size_t lds2 = (size_t)64 * XS * 2 + (size_t)(NH * TC + TC + 64 * NH) * 4;
  k_pass2<<<dim3(NB, 16), 256, lds2, stream>>>(x, wqp, bq, ksum, u, bp, out);
}
